// Attention_2156073582718
// MI455X (gfx1250) — matrix operand order repaired and checked by reading, not compiled, not run on hardware
//
#include <hip/hip_runtime.h>

// ---------------------------------------------------------------------------
// Single-head causal attention, B=4 T=4096 C=768, fp32 in/out.
// f16 WMMA (v_wmma_f32_16x16x32_f16) with f32 accumulation everywhere.
// Q tile staged to LDS via the CDNA5 async copy path (ASYNCcnt).
// ---------------------------------------------------------------------------

typedef __attribute__((ext_vector_type(16))) _Float16 v16h;
typedef __attribute__((ext_vector_type(8)))  _Float16 v8h;
typedef __attribute__((ext_vector_type(8)))  float    v8f;

#define DEVINL __device__ __forceinline__

constexpr int Bb   = 4;
constexpr int T    = 4096;
constexpr int C    = 768;
constexpr int BT   = Bb * T;          // 16384
constexpr int NBTC = Bb * T * C;      // 12,582,912

// ---- WMMA fragment loaders (contiguous k = 16h + s, same order in A and B) ------------------
// A 16x32 f16: lane m = lane%16 ; kb = lane<16 ? 0 : 16 ;
//   elements 0..15 = K kb..kb+15 (contiguous)
DEVINL v16h load_a_frag(const _Float16* base, int ld, int m0, int c0) {
  const int lane = threadIdx.x & 31;
  const int m  = m0 + (lane & 15);
  const int kb = (lane & 16) ? 16 : 0;
  const _Float16* p = base + (size_t)m * ld + (c0 + kb);
  v8h lo = *(const v8h*)p;
  v8h hi = *(const v8h*)(p + 8);
  v16h r;
#pragma unroll
  for (int i = 0; i < 8; ++i) { r[i] = lo[i]; r[8 + i] = hi[i]; }
  return r;
}

// B 32x16 f16 (stored as rows of N with contiguous K): lane n = lane%16 ;
//   kb = lane<16 ? 0 : 16 ; elements 0..15 = K kb..kb+15 (contiguous 32B)
DEVINL v16h load_b_frag(const _Float16* base, int ld, int n0, int c0) {
  const int lane = threadIdx.x & 31;
  const int n  = n0 + (lane & 15);
  const int kb = (lane & 16) ? 16 : 0;
  return *(const v16h*)(base + (size_t)n * ld + (c0 + kb));
}

DEVINL v8f wmma_f16(v16h a, v16h b, v8f c) {
  return __builtin_amdgcn_wmma_f32_16x16x32_f16(false, a, false, b,
                                                (short)0, c, false, false);
}

// Async global -> LDS copy of 16B per lane (GLOBAL_LOAD_ASYNC_TO_LDS_B128).
// lds generic pointer: low 32 bits are the wave-relative LDS byte address
// (ISA 10.2).
DEVINL void async_copy_b128(void* lds_dst, const void* gsrc) {
  unsigned ldsoff = (unsigned)(uintptr_t)lds_dst;
  asm volatile("global_load_async_to_lds_b128 %0, %1, off"
               :: "v"(ldsoff), "v"(gsrc) : "memory");
}
DEVINL void async_wait0() {
  asm volatile("s_wait_asynccnt 0x0" ::: "memory");
}

// ---- prep kernels ----------------------------------------------------------
__global__ void prep_input(const float* __restrict__ x,
                           _Float16* __restrict__ x16) {
  int i = blockIdx.x * 256 + threadIdx.x;
  if (i < NBTC) x16[i] = (_Float16)x[i];
}

// Wt16[z][n*C + k] = W_z[k*C + n]  (f16, transposed so B-frags are contiguous)
__global__ void prep_weights(const float* __restrict__ Wq,
                             const float* __restrict__ Wk,
                             const float* __restrict__ Wv,
                             _Float16* __restrict__ Wt16) {
  int i = blockIdx.x * 256 + threadIdx.x;
  if (i >= 3 * C * C) return;
  int z = i / (C * C);
  int r = i - z * C * C;
  int n = r / C, k = r - n * C;
  const float* W = (z == 0) ? Wq : (z == 1) ? Wk : Wv;
  Wt16[i] = (_Float16)W[k * C + n];
}

// ---- fused QKV projection GEMM --------------------------------------------
// out[m][n] = X[m][:] . W[:][n] + bias[n] ; M=BT, N=C, K=C.
// grid = (BT/128, C/64, 3) ; block = 256 (8 waves, each 32x32 of the tile).
// z==2 (V) stores transposed: Vt[b][n][t].
__global__ __launch_bounds__(256)
void qkv_gemm(const _Float16* __restrict__ X, const _Float16* __restrict__ Wt,
              const float* __restrict__ bq, const float* __restrict__ bk,
              const float* __restrict__ bv,
              _Float16* __restrict__ Qo, _Float16* __restrict__ Ko,
              _Float16* __restrict__ Vt) {
  const int z    = blockIdx.z;
  const _Float16* W = Wt + (size_t)z * C * C;
  const float* bias = (z == 0) ? bq : (z == 1) ? bk : bv;

  const int w    = threadIdx.x >> 5;
  const int lane = threadIdx.x & 31;
  const int mblk = blockIdx.x * 128 + (w >> 1) * 32;
  const int nblk = blockIdx.y * 64 + (w & 1) * 32;

  v8f acc[2][2] = {};
  for (int c0 = 0; c0 < C; c0 += 32) {
    v16h a0 = load_a_frag(X, C, mblk,      c0);
    v16h a1 = load_a_frag(X, C, mblk + 16, c0);
    v16h b0 = load_b_frag(W, C, nblk,      c0);
    v16h b1 = load_b_frag(W, C, nblk + 16, c0);
    acc[0][0] = wmma_f16(a0, b0, acc[0][0]);
    acc[0][1] = wmma_f16(a0, b1, acc[0][1]);
    acc[1][0] = wmma_f16(a1, b0, acc[1][0]);
    acc[1][1] = wmma_f16(a1, b1, acc[1][1]);
  }

  const int rbase = (lane & 16) ? 8 : 0;
#pragma unroll
  for (int mi = 0; mi < 2; ++mi)
#pragma unroll
    for (int ni = 0; ni < 2; ++ni) {
      const int n = nblk + ni * 16 + (lane & 15);
      const float bsv = bias[n];
#pragma unroll
      for (int r = 0; r < 8; ++r) {
        const int m = mblk + mi * 16 + r + rbase;
        const float val = acc[mi][ni][r] + bsv;
        if (z == 0)      Qo[(size_t)m * C + n] = (_Float16)val;
        else if (z == 1) Ko[(size_t)m * C + n] = (_Float16)val;
        else {
          const int bi = m >> 12, t = m & (T - 1);
          Vt[((size_t)bi * C + n) * T + t] = (_Float16)val;
        }
      }
    }
}

// ---- flash attention -------------------------------------------------------
// grid = (T/32, B), block = 256 (8 waves).
// Block owns 32 queries (async-staged into LDS once). Key tiles of 128;
// wave w computes the 16-key strip S[:, w*16..+16] (full C contraction),
// then accumulates O over a 96-channel strip (12 f32 WMMA accs in regs).
constexpr int QLD = 776;   // 768 + 8 pad: row stride 1552B = 388 dw ≡ 4 mod 64
constexpr int SLD = 132;   // f32 pad
constexpr int PLD = 136;   // f16 pad: row stride 272B = 68 dw ≡ 4 mod 64

__global__ __launch_bounds__(256)
void attn(const _Float16* __restrict__ Qg, const _Float16* __restrict__ Kg,
          const _Float16* __restrict__ Vtg, float* __restrict__ Out) {
  __shared__ __align__(32) _Float16 sQ[32][QLD];
  __shared__ float    sS[32][SLD];
  __shared__ __align__(32) _Float16 sP[32][PLD];
  __shared__ float    sM[32], sL[32], sAlpha[32];

  const int b    = blockIdx.y;
  const int qt   = blockIdx.x * 32;
  const int w    = threadIdx.x >> 5;
  const int lane = threadIdx.x & 31;
  const int rbase = (lane & 16) ? 8 : 0;
  const float scale = 0.03608439182435161f;   // 1/sqrt(768)

  const _Float16* Qb = Qg  + (size_t)b * T * C;
  const _Float16* Kb = Kg  + (size_t)b * T * C;
  const _Float16* Vb = Vtg + (size_t)b * C * T;   // [c][t]

  // Stage the 32x768 Q tile into LDS via the async copy path (16B/lane,
  // 3072 chunks = 12 full-EXEC rounds of 256 lanes).
  for (int ch = threadIdx.x; ch < 32 * (C / 8); ch += 256) {
    const int row = ch / (C / 8);
    const int off = (ch - row * (C / 8)) * 8;
    async_copy_b128(&sQ[row][off], Qb + (size_t)(qt + row) * C + off);
  }
  if (threadIdx.x < 32) { sM[threadIdx.x] = -1e30f; sL[threadIdx.x] = 0.0f; }

  v8f o[2][6];
#pragma unroll
  for (int mi = 0; mi < 2; ++mi)
#pragma unroll
    for (int ni = 0; ni < 6; ++ni) o[mi][ni] = (v8f){};

  const int cstrip = w * 96;
  const int jmax = (qt + 31) >> 7;

  async_wait0();     // Q tile landed in LDS (this wave's share)

  for (int j = 0; j <= jmax; ++j) {
    const int keyt = j * 128;
    const int key0 = keyt + w * 16;
    __syncthreads();   // sQ staged / sS,sP free from previous iteration

    // Prefetch next key tile's K strip (global_prefetch_b8).
    if (j < jmax) {
      const char* nk =
          (const char*)(Kb + (size_t)(keyt + 128 + w * 16 + (lane & 15)) * C) +
          (lane >> 4) * 768;
#pragma unroll
      for (int i = 0; i < 6; ++i) __builtin_prefetch(nk + i * 128, 0, 3);
    }

    // ---- S = Q K^T (this wave's 16-key strip, both 16-query subtiles) ----
    v8f s0 = (v8f){}, s1 = (v8f){};
    for (int c0 = 0; c0 < C; c0 += 32) {
      v16h kf  = load_b_frag(Kb, C, key0, c0);
      v16h qa0 = load_a_frag((const _Float16*)sQ, QLD, 0,  c0);
      v16h qa1 = load_a_frag((const _Float16*)sQ, QLD, 16, c0);
      s0 = wmma_f16(qa0, kf, s0);
      s1 = wmma_f16(qa1, kf, s1);
    }
    {
      const int n   = key0 + (lane & 15);          // absolute key
      const int col = w * 16 + (lane & 15);        // column in sS
#pragma unroll
      for (int r = 0; r < 8; ++r) {
        const int q0 = qt + r + rbase;             // subtile mi=0
        sS[r + rbase][col]      = (n <= q0)      ? s0[r] * scale : -1e30f;
        sS[r + rbase + 16][col] = (n <= q0 + 16) ? s1[r] * scale : -1e30f;
      }
    }
    __syncthreads();

    // ---- online softmax over the 128-key tile (8 threads per row) --------
    {
      const int row = threadIdx.x >> 3;
      const int seg = threadIdx.x & 7;
      float vals[16];
      float mx = -1e30f;
#pragma unroll
      for (int i = 0; i < 16; ++i) {
        vals[i] = sS[row][seg * 16 + i];
        mx = fmaxf(mx, vals[i]);
      }
#pragma unroll
      for (int off = 1; off < 8; off <<= 1) mx = fmaxf(mx, __shfl_xor(mx, off, 32));
      const float m_old = sM[row];
      const float m_new = fmaxf(m_old, mx);
      float sum = 0.0f;
#pragma unroll
      for (int i = 0; i < 16; ++i) {
        const float e = __expf(vals[i] - m_new);
        sP[row][seg * 16 + i] = (_Float16)e;
        sum += e;
      }
#pragma unroll
      for (int off = 1; off < 8; off <<= 1) sum += __shfl_xor(sum, off, 32);
      if (seg == 0) {
        const float alpha = __expf(m_old - m_new);
        sAlpha[row] = alpha;
        sL[row] = sL[row] * alpha + sum;
        sM[row] = m_new;
      }
    }
    __syncthreads();

    // ---- rescale running O by alpha(row) ---------------------------------
    float al[2][8];
#pragma unroll
    for (int mi = 0; mi < 2; ++mi)
#pragma unroll
      for (int r = 0; r < 8; ++r) al[mi][r] = sAlpha[mi * 16 + r + rbase];
#pragma unroll
    for (int mi = 0; mi < 2; ++mi)
#pragma unroll
      for (int ni = 0; ni < 6; ++ni)
#pragma unroll
        for (int r = 0; r < 8; ++r) o[mi][ni][r] *= al[mi][r];

    // ---- O += P (32x128) x V (128 x 96-strip) ----------------------------
    for (int kk = 0; kk < 4; ++kk) {
      v16h p0 = load_a_frag((const _Float16*)sP, PLD, 0,  kk * 32);
      v16h p1 = load_a_frag((const _Float16*)sP, PLD, 16, kk * 32);
#pragma unroll
      for (int ni = 0; ni < 6; ++ni) {
        v16h vf = load_b_frag(Vb, T, cstrip + ni * 16, keyt + kk * 32);
        o[0][ni] = wmma_f16(p0, vf, o[0][ni]);
        o[1][ni] = wmma_f16(p1, vf, o[1][ni]);
      }
    }
  }

  // ---- normalize and write fp32 output -----------------------------------
  float linv[2][8];
#pragma unroll
  for (int mi = 0; mi < 2; ++mi)
#pragma unroll
    for (int r = 0; r < 8; ++r) linv[mi][r] = 1.0f / sL[mi * 16 + r + rbase];

#pragma unroll
  for (int mi = 0; mi < 2; ++mi)
#pragma unroll
    for (int ni = 0; ni < 6; ++ni) {
      const int c = cstrip + ni * 16 + (lane & 15);
#pragma unroll
      for (int r = 0; r < 8; ++r) {
        const int q = qt + mi * 16 + r + rbase;
        Out[((size_t)b * T + q) * C + c] = o[mi][ni][r] * linv[mi][r];
      }
    }
}

// ---------------------------------------------------------------------------
extern "C" void kernel_launch(void* const* d_in, const int* in_sizes, int n_in,
                              void* d_out, int out_size, void* d_ws, size_t ws_size,
                              hipStream_t stream) {
  const float* x  = (const float*)d_in[0];
  const float* Wq = (const float*)d_in[1];
  const float* bq = (const float*)d_in[2];
  const float* Wk = (const float*)d_in[3];
  const float* bk = (const float*)d_in[4];
  const float* Wv = (const float*)d_in[5];
  const float* bv = (const float*)d_in[6];
  float* out = (float*)d_out;

  _Float16* X16  = (_Float16*)d_ws;
  _Float16* Q16  = X16  + NBTC;
  _Float16* K16  = Q16  + NBTC;
  _Float16* V16t = K16  + NBTC;
  _Float16* Wt16 = V16t + NBTC;   // 3*C*C f16

  prep_input  <<<(NBTC + 255) / 256, 256, 0, stream>>>(x, X16);
  prep_weights<<<(3 * C * C + 255) / 256, 256, 0, stream>>>(Wq, Wk, Wv, Wt16);
  qkv_gemm    <<<dim3(BT / 128, C / 64, 3), 256, 0, stream>>>(
      X16, Wt16, bq, bk, bv, Q16, K16, V16t);
  attn        <<<dim3(T / 32, Bb), 256, 0, stream>>>(Q16, K16, V16t, out);
}
